// FermiNet_43482248905402
// MI455X (gfx1250) — matrix operand order repaired and checked by reading, not compiled, not run on hardware
//
#include <hip/hip_runtime.h>
#include <hip/hip_bf16.h>
#include <math.h>

#define B_TOT   4096
#define NE      16
#define DIM     256
#define NO      128
#define NIONS   4

typedef __attribute__((ext_vector_type(16))) __bf16         v16bf;
typedef __attribute__((ext_vector_type(2)))  __bf16         v2bf;
typedef __attribute__((ext_vector_type(16))) unsigned short v16u;
typedef __attribute__((ext_vector_type(8)))  unsigned int   v8u;
typedef __attribute__((ext_vector_type(8)))  float          v8f;
typedef __attribute__((ext_vector_type(4)))  float          v4f;

// float -> bf16 round-to-nearest-even (scalar, used in prep kernel only)
__device__ __forceinline__ unsigned short f2bf(float f) {
    union { float f; unsigned u; } x; x.f = f;
    unsigned u = x.u;
    u += 0x7FFFu + ((u >> 16) & 1u);
    return (unsigned short)(u >> 16);
}

union afrag_u {
    v16bf b;
    v2bf  p[8];
    v16u  u;
    v8u   d;
};

// Pack two f32 -> one dword of two bf16 (round-half-up + v_perm_b32 byte pick).
__device__ __forceinline__ unsigned pack2bf(float lo, float hi) {
    unsigned a = __float_as_uint(lo) + 0x8000u;
    unsigned b = __float_as_uint(hi) + 0x8000u;
    // v_perm_b32: byte indices 0-3 select from S1 (a), 4-7 from S0 (b).
    // result = { b[31:16], a[31:16] }
    return __builtin_amdgcn_perm(b, a, 0x07060302u);
}

// Pack 16 floats (four v4f) into a v16bf A-fragment.
__device__ __forceinline__ void pack16(afrag_u& af, v4f f0, v4f f1, v4f f2, v4f f3) {
#if __has_builtin(__builtin_amdgcn_cvt_pk_bf16_f32)
    af.p[0] = __builtin_amdgcn_cvt_pk_bf16_f32(f0[0], f0[1]);
    af.p[1] = __builtin_amdgcn_cvt_pk_bf16_f32(f0[2], f0[3]);
    af.p[2] = __builtin_amdgcn_cvt_pk_bf16_f32(f1[0], f1[1]);
    af.p[3] = __builtin_amdgcn_cvt_pk_bf16_f32(f1[2], f1[3]);
    af.p[4] = __builtin_amdgcn_cvt_pk_bf16_f32(f2[0], f2[1]);
    af.p[5] = __builtin_amdgcn_cvt_pk_bf16_f32(f2[2], f2[3]);
    af.p[6] = __builtin_amdgcn_cvt_pk_bf16_f32(f3[0], f3[1]);
    af.p[7] = __builtin_amdgcn_cvt_pk_bf16_f32(f3[2], f3[3]);
#else
    af.d[0] = pack2bf(f0[0], f0[1]);
    af.d[1] = pack2bf(f0[2], f0[3]);
    af.d[2] = pack2bf(f1[0], f1[1]);
    af.d[3] = pack2bf(f1[2], f1[3]);
    af.d[4] = pack2bf(f2[0], f2[1]);
    af.d[5] = pack2bf(f2[2], f2[3]);
    af.d[6] = pack2bf(f3[0], f3[1]);
    af.d[7] = pack2bf(f3[2], f3[3]);
#endif
}

// Repack W (D x NO, f32 row-major) into the per-lane WMMA B-fragment layout, bf16.
// Layout: [spin][kb(8)][nt(8)][lane(32)][j(16)]  (contiguous 32B per lane per fragment)
//   n = nt*16 + (lane&15);  k = kb*32 + 16*(j>=8) + 8*(lane>=16) + (j&7)
__global__ __launch_bounds__(256) void prep_w_kernel(const float* __restrict__ Wu,
                                                     const float* __restrict__ Wd,
                                                     unsigned short* __restrict__ wsW) {
    int tid  = blockIdx.x * 256 + threadIdx.x;   // 65536 total
    int j    =  tid        & 15;
    int lane = (tid >> 4)  & 31;
    int nt   = (tid >> 9)  & 7;
    int kb   = (tid >> 12) & 7;
    int spin = (tid >> 15) & 1;
    int n = nt * 16 + (lane & 15);
    int k = kb * 32 + ((j >> 3) << 4) + ((lane >> 4) << 3) + (j & 7);
    const float* W = spin ? Wd : Wu;
    wsW[tid] = f2bf(W[k * NO + n]);
}

// Fused: GEMM (WMMA bf16) -> bias+envelope -> 8x8 slogdet per lane -> signed logsumexp.
// Block = 128 threads = 4 waves. Wave w: spin = w&1, sample-pair = w>>1.
// Each wave: M-tile = 16 rows (2 samples x 8 electrons of its spin), full N=128.
__global__ __launch_bounds__(128) void ferminet_fused_kernel(
    const float* __restrict__ stream,
    const float* __restrict__ r_ei,
    const float* __restrict__ b_up, const float* __restrict__ pi_up, const float* __restrict__ a_up,
    const float* __restrict__ b_dn, const float* __restrict__ pi_dn, const float* __restrict__ a_dn,
    const unsigned short* __restrict__ wsW,
    float* __restrict__ out)
{
    __shared__ float ldsDist[4][2][8][4];   // [wave][sample][electron][ion]
    __shared__ float ldsOrb[4][2120];       // per-wave orbitals: s*1060 + e*132 + o (padded)
    __shared__ float ldsSign[2][4][16];     // [spin][sampleInBlock][det]
    __shared__ float ldsLog [2][4][16];

    const int lane = threadIdx.x & 31;
    const int w    = threadIdx.x >> 5;
    const int spin = w & 1;
    const int pair = w >> 1;
    const int b0   = blockIdx.x * 4 + pair * 2;   // wave covers samples b0, b0+1

    // ---- electron-ion distances (cooperative: 2 values per lane) ----
    #pragma unroll
    for (int t = 0; t < 2; ++t) {
        int idx = lane * 2 + t;                 // 0..63
        int s_  = idx >> 5;
        int el  = (idx >> 2) & 7;
        int io  = idx & 3;
        int e   = spin * 8 + el;
        int base = (((b0 + s_) * NE + e) * NIONS + io) * 3;
        float x = r_ei[base], y = r_ei[base + 1], z = r_ei[base + 2];
        ldsDist[w][s_][el][io] = sqrtf(x * x + y * y + z * z);
    }
    __syncthreads();

    // ---- GEMM: acc[nt] (16x16 f32) over K=256 in 8 steps of 32 (bf16) ----
    v8f acc[8] = {};
    const int m     = lane & 15;                      // M-row within tile
    const int bsamp = b0 + (m >> 3);
    const int e_row = spin * 8 + (m & 7);
    const float* arow = stream + ((long)bsamp * NE + e_row) * DIM;
    const int khalf = (lane >> 4) * 8;                // A-frag K split per ISA layout
    const unsigned short* wsp = wsW + spin * (8 * 8 * 512) + lane * 16;

    #pragma unroll
    for (int kb = 0; kb < 8; ++kb) {
        const float* p = arow + kb * 32 + khalf;
        v4f f0 = *(const v4f*)(p);
        v4f f1 = *(const v4f*)(p + 4);
        v4f f2 = *(const v4f*)(p + 16);
        v4f f3 = *(const v4f*)(p + 20);
        afrag_u af;
        pack16(af, f0, f1, f2, f3);
        const unsigned short* wkb = wsp + kb * (8 * 512);
        #pragma unroll
        for (int nt = 0; nt < 8; ++nt) {
            v16bf bfrag = *(const v16bf*)(wkb + nt * 512);
            acc[nt] = __builtin_amdgcn_wmma_f32_16x16x32_bf16(
                false, af.b, false, bfrag, (short)0, acc[nt], false, false);
        }
    }

    // ---- bias + envelope, store orbitals to LDS ----
    // C layout: lanes 0-15 -> rows 0-7 (sample 0), lanes 16-31 -> rows 8-15 (sample 1);
    // column N = lane&15, row M-within-sample = vgpr index v.
    const int s_   = lane >> 4;
    const int ocol = lane & 15;
    float dd[8][4];
    #pragma unroll
    for (int ee = 0; ee < 8; ++ee)
        #pragma unroll
        for (int io = 0; io < 4; ++io)
            dd[ee][io] = ldsDist[w][s_][ee][io];

    const float* bb  = spin ? b_dn  : b_up;
    const float* pp  = spin ? pi_dn : pi_up;
    const float* aav = spin ? a_dn  : a_up;
    float* orbw = &ldsOrb[w][0];

    #pragma unroll
    for (int nt = 0; nt < 8; ++nt) {
        int o = nt * 16 + ocol;
        float bias = bb[o];
        float p0 = pp[o], p1 = pp[NO + o], p2 = pp[2 * NO + o], p3 = pp[3 * NO + o];
        float a0 = aav[o], a1 = aav[NO + o], a2 = aav[2 * NO + o], a3 = aav[3 * NO + o];
        #pragma unroll
        for (int v = 0; v < 8; ++v) {
            float env = p0 * __expf(-a0 * dd[v][0])
                      + p1 * __expf(-a1 * dd[v][1])
                      + p2 * __expf(-a2 * dd[v][2])
                      + p3 * __expf(-a3 * dd[v][3]);
            orbw[s_ * 1060 + v * 132 + o] = (acc[nt][v] + bias) * env;
        }
    }
    __syncthreads();

    // ---- one 8x8 slogdet per lane (32 dets = 2 samples x 16 dets per wave) ----
    const int sdet = lane >> 4;
    const int det  = lane & 15;
    float A[8][8];
    #pragma unroll
    for (int r = 0; r < 8; ++r)
        #pragma unroll
        for (int c = 0; c < 8; ++c)
            A[r][c] = orbw[sdet * 1060 + r * 132 + det * 8 + c];

    float sgn = 1.0f, ld = 0.0f;
    #pragma unroll
    for (int k = 0; k < 8; ++k) {
        // partial pivoting via static conditional row swaps (keeps A in VGPRs)
        #pragma unroll
        for (int i = k + 1; i < 8; ++i) {
            bool sw = __builtin_fabsf(A[i][k]) > __builtin_fabsf(A[k][k]);
            #pragma unroll
            for (int j2 = k; j2 < 8; ++j2) {
                float tk = A[k][j2], ti = A[i][j2];
                A[k][j2] = sw ? ti : tk;
                A[i][j2] = sw ? tk : ti;
            }
            sgn = sw ? -sgn : sgn;
        }
        float piv = A[k][k];
        ld += __logf(__builtin_fabsf(piv));
        sgn = (piv < 0.0f) ? -sgn : sgn;
        float inv = 1.0f / piv;
        #pragma unroll
        for (int i = k + 1; i < 8; ++i) {
            float f = A[i][k] * inv;
            #pragma unroll
            for (int j2 = k + 1; j2 < 8; ++j2)
                A[i][j2] = fmaf(-f, A[k][j2], A[i][j2]);
        }
    }
    ldsSign[spin][pair * 2 + sdet][det] = sgn;
    ldsLog [spin][pair * 2 + sdet][det] = ld;
    __syncthreads();

    // ---- signed logsumexp over 16 determinants per sample ----
    if (threadIdx.x < 64) {
        int sb = threadIdx.x >> 4;    // sample in block (0..3)
        int d  = threadIdx.x & 15;
        float s2 = ldsSign[0][sb][d] * ldsSign[1][sb][d];
        float v2 = ldsLog [0][sb][d] + ldsLog [1][sb][d];
        float mx = v2;
        #pragma unroll
        for (int off = 8; off >= 1; off >>= 1)
            mx = fmaxf(mx, __shfl_xor(mx, off, 16));
        float tsum = s2 * __expf(v2 - mx);
        #pragma unroll
        for (int off = 8; off >= 1; off >>= 1)
            tsum += __shfl_xor(tsum, off, 16);
        if (d == 0)
            out[blockIdx.x * 4 + sb] = mx + __logf(__builtin_fabsf(tsum));
    }
}

extern "C" void kernel_launch(void* const* d_in, const int* in_sizes, int n_in,
                              void* d_out, int out_size, void* d_ws, size_t ws_size,
                              hipStream_t stream) {
    const float* stream_1e = (const float*)d_in[0];
    const float* r_ei      = (const float*)d_in[1];
    const float* W_up      = (const float*)d_in[2];
    const float* b_up      = (const float*)d_in[3];
    const float* pi_up     = (const float*)d_in[4];
    const float* a_up      = (const float*)d_in[5];
    const float* W_dn      = (const float*)d_in[6];
    const float* b_dn      = (const float*)d_in[7];
    const float* pi_dn     = (const float*)d_in[8];
    const float* a_dn      = (const float*)d_in[9];
    float* out = (float*)d_out;
    unsigned short* wsW = (unsigned short*)d_ws;   // 131072 bytes used

    prep_w_kernel<<<256, 256, 0, stream>>>(W_up, W_dn, wsW);
    ferminet_fused_kernel<<<B_TOT / 4, 128, 0, stream>>>(
        stream_1e, r_ei, b_up, pi_up, a_up, b_dn, pi_dn, a_dn, wsW, out);
}
